// InternLM2DecoderLayer_28827820491094
// MI455X (gfx1250) — matrix operand order repaired and checked by reading, not compiled, not run on hardware
//
#include <hip/hip_runtime.h>

// ---------------------------------------------------------------------------
// InternLM2 decoder layer for MI455X (gfx1250), wave32 + WMMA bf16 +
// async global->LDS DMA (GLOBAL_LOAD_ASYNC_TO_LDS_B128, ASYNCcnt).
// ---------------------------------------------------------------------------

#define T_TOK 2048
#define HIDN  4096
#define NH    32
#define NKV   8
#define HD    128
#define QKVD  ((NH + 2 * NKV) * HD)   // 6144
#define INTER 14336
#define RMS_EPS 1e-5f

typedef __attribute__((ext_vector_type(16))) __bf16 v16bf;
typedef __attribute__((ext_vector_type(8)))  __bf16 v8bf;
typedef __attribute__((ext_vector_type(4)))  __bf16 v4bf;
typedef __attribute__((ext_vector_type(8)))  float  v8f;
typedef __attribute__((ext_vector_type(4)))  float  v4f;

static __device__ __forceinline__ v8f wmma_bf16(v16bf a, v16bf b, v8f c) {
  return __builtin_amdgcn_wmma_f32_16x16x32_bf16(
      false, a, false, b, (short)0, c, false, false);
}

// A operand (16x32, 16-bit) as loaded here, contiguous k order (B matches): lane L -> M = L%16, half = L/16.
// element e -> K = e + half*16.
static __device__ __forceinline__ v16bf load_a_frag(const __bf16* row, int hf) {
  union { v16bf v; v8bf h[2]; } u;
  u.h[0] = *(const v8bf*)(row + hf * 16);
  u.h[1] = *(const v8bf*)(row + hf * 16 + 8);
  return u.v;
}

// B operand (32x16, 16-bit) as loaded here, same k order as A: lane L -> N = L%16, half = L/16, K = e + 16*half.
static __device__ __forceinline__ v16bf load_b_frag(const __bf16* p) {
  union { v16bf v; v8bf h[2]; } u;
  u.h[0] = *(const v8bf*)(p);
  u.h[1] = *(const v8bf*)(p + 8);
  return u.v;
}

// Async 16B copy global -> LDS, tracked by ASYNCcnt (CDNA5 ISA 15.18.3 op 98).
// Generic pointer to __shared__ has the LDS byte offset in its low 32 bits
// (flat LDS aperture).
static __device__ __forceinline__ void async_copy_b128(void* lds_ptr,
                                                       const void* gptr) {
  unsigned l = (unsigned)(uintptr_t)lds_ptr;
  asm volatile("global_load_async_to_lds_b128 %0, %1, off"
               :: "v"(l), "v"(gptr) : "memory");
}
static __device__ __forceinline__ void wait_async0() {
  asm volatile("s_wait_asynccnt 0x0" ::: "memory");
}

// ---------------------------------------------------------------------------
// fp32 -> bf16 (weight conversion), vectorized 4-wide
// ---------------------------------------------------------------------------
__global__ void cvt_bf16_kernel(const float* __restrict__ in,
                                __bf16* __restrict__ out, size_t n4) {
  size_t i = (size_t)blockIdx.x * blockDim.x + threadIdx.x;
  size_t stride = (size_t)gridDim.x * blockDim.x;
  for (; i < n4; i += stride) {
    v4f x = *(const v4f*)(in + 4 * i);
    v4bf y;
    y[0] = (__bf16)x[0]; y[1] = (__bf16)x[1];
    y[2] = (__bf16)x[2]; y[3] = (__bf16)x[3];
    *(v4bf*)(out + 4 * i) = y;
  }
}

// ---------------------------------------------------------------------------
// RMSNorm row kernel: fp32 [T, HIDN] -> bf16 [T, HIDN]
// ---------------------------------------------------------------------------
__global__ __launch_bounds__(256) void rmsnorm_bf16_kernel(
    const float* __restrict__ x, const float* __restrict__ w,
    __bf16* __restrict__ out) {
  __shared__ float red[8];
  int t = blockIdx.x;
  const float* xr = x + (size_t)t * HIDN;
  float ss = 0.f;
  for (int i = threadIdx.x; i < HIDN; i += 256) {
    float v = xr[i];
    ss += v * v;
  }
  for (int off = 16; off; off >>= 1) ss += __shfl_xor(ss, off, 32);
  if ((threadIdx.x & 31) == 0) red[threadIdx.x >> 5] = ss;
  __syncthreads();
  float tot = 0.f;
#pragma unroll
  for (int i = 0; i < 8; ++i) tot += red[i];
  float rinv = rsqrtf(tot / (float)HIDN + RMS_EPS);
  for (int i = threadIdx.x; i < HIDN; i += 256)
    out[(size_t)t * HIDN + i] = (__bf16)(xr[i] * rinv * w[i]);
}

// ---------------------------------------------------------------------------
// WMMA GEMM: C[M,N] = A[M,Kd]*B[N,Kd]^T (+resid). Block tile 128x128, 8 waves,
// wave tile 64x32 (4x2 accumulators -> 8 WMMAs/K-step). K-step 32.
// Double-buffered LDS filled by async global->LDS DMA.
// ---------------------------------------------------------------------------
template <int EPI>
__global__ __launch_bounds__(256) void wmma_gemm_kernel(
    const __bf16* __restrict__ A, const __bf16* __restrict__ B,
    const float* __restrict__ resid, float* __restrict__ C,
    int M, int N, int Kd) {
  __shared__ __bf16 As[2][128][40];
  __shared__ __bf16 Bs[2][128][40];
  int tid = threadIdx.x;
  int lane = tid & 31, wid = tid >> 5;
  int wm = wid >> 2, wn = wid & 3;   // wave: rows wm*64, cols wn*32
  int m0 = blockIdx.y * 128, n0 = blockIdx.x * 128;
  int mlane = lane & 15, hf = lane >> 4;

  v8f acc[4][2];
#pragma unroll
  for (int mt = 0; mt < 4; ++mt)
#pragma unroll
    for (int nt = 0; nt < 2; ++nt) acc[mt][nt] = (v8f){};

  // stage: 128 rows x 32 cols per matrix = 512 16B-chunks -> 2 per thread
  auto stage = [&](int buf, int k0) {
#pragma unroll
    for (int i = 0; i < 2; ++i) {
      int c = i * 256 + tid;
      int r = c >> 2, col = (c & 3) * 8;
      async_copy_b128(&As[buf][r][col], &A[(size_t)(m0 + r) * Kd + k0 + col]);
      async_copy_b128(&Bs[buf][r][col], &B[(size_t)(n0 + r) * Kd + k0 + col]);
    }
  };

  stage(0, 0);
  int nkb = Kd >> 5;
  for (int kb = 0; kb < nkb; ++kb) {
    int buf = kb & 1;
    wait_async0();
    __syncthreads();
    if (kb + 1 < nkb) stage(buf ^ 1, (kb + 1) << 5);

    v16bf af[4], bfr[2];
#pragma unroll
    for (int mt = 0; mt < 4; ++mt)
      af[mt] = load_a_frag(&As[buf][wm * 64 + mt * 16 + mlane][0], hf);
#pragma unroll
    for (int nt = 0; nt < 2; ++nt)
      bfr[nt] = load_b_frag(&Bs[buf][wn * 32 + nt * 16 + mlane][hf * 16]);
#pragma unroll
    for (int mt = 0; mt < 4; ++mt)
#pragma unroll
      for (int nt = 0; nt < 2; ++nt)
        acc[mt][nt] = wmma_bf16(af[mt], bfr[nt], acc[mt][nt]);
    __syncthreads();
  }

#pragma unroll
  for (int mt = 0; mt < 4; ++mt)
#pragma unroll
    for (int nt = 0; nt < 2; ++nt)
#pragma unroll
      for (int r = 0; r < 8; ++r) {
        int mg = m0 + wm * 64 + mt * 16 + r + 8 * hf;
        int ng = n0 + wn * 32 + nt * 16 + mlane;
        size_t i0 = (size_t)mg * N + ng;
        float v = acc[mt][nt][r];
        if (EPI == 1) v += resid[i0];
        C[i0] = v;
      }
}

// ---------------------------------------------------------------------------
// Fused gated GEMM: out_bf16 = silu(A*W1^T) * (A*W3^T). Block tile 64x128,
// wave tile 32x32 per B (2x2x2 accumulators). Async double-buffered LDS.
// ---------------------------------------------------------------------------
__global__ __launch_bounds__(256) void wmma_gemm_gated_kernel(
    const __bf16* __restrict__ A, const __bf16* __restrict__ B1,
    const __bf16* __restrict__ B3, __bf16* __restrict__ C,
    int M, int N, int Kd) {
  __shared__ __bf16 As[2][64][40];
  __shared__ __bf16 B1s[2][128][40];
  __shared__ __bf16 B3s[2][128][40];
  int tid = threadIdx.x;
  int lane = tid & 31, wid = tid >> 5;
  int wm = wid >> 2, wn = wid & 3;   // rows wm*32, cols wn*32
  int m0 = blockIdx.y * 64, n0 = blockIdx.x * 128;
  int mlane = lane & 15, hf = lane >> 4;

  v8f acc1[2][2], acc3[2][2];
#pragma unroll
  for (int mt = 0; mt < 2; ++mt)
#pragma unroll
    for (int nt = 0; nt < 2; ++nt) { acc1[mt][nt] = (v8f){}; acc3[mt][nt] = (v8f){}; }

  auto stage = [&](int buf, int k0) {
    {  // A: 64x32 = 256 chunks -> 1 per thread
      int r = tid >> 2, col = (tid & 3) * 8;
      async_copy_b128(&As[buf][r][col], &A[(size_t)(m0 + r) * Kd + k0 + col]);
    }
#pragma unroll
    for (int i = 0; i < 2; ++i) {  // B1/B3: 128x32 = 512 chunks -> 2 per thread
      int c = i * 256 + tid;
      int r = c >> 2, col = (c & 3) * 8;
      async_copy_b128(&B1s[buf][r][col], &B1[(size_t)(n0 + r) * Kd + k0 + col]);
      async_copy_b128(&B3s[buf][r][col], &B3[(size_t)(n0 + r) * Kd + k0 + col]);
    }
  };

  stage(0, 0);
  int nkb = Kd >> 5;
  for (int kb = 0; kb < nkb; ++kb) {
    int buf = kb & 1;
    wait_async0();
    __syncthreads();
    if (kb + 1 < nkb) stage(buf ^ 1, (kb + 1) << 5);

    v16bf af[2], b1f[2], b3f[2];
#pragma unroll
    for (int mt = 0; mt < 2; ++mt)
      af[mt] = load_a_frag(&As[buf][wm * 32 + mt * 16 + mlane][0], hf);
#pragma unroll
    for (int nt = 0; nt < 2; ++nt) {
      b1f[nt] = load_b_frag(&B1s[buf][wn * 32 + nt * 16 + mlane][hf * 16]);
      b3f[nt] = load_b_frag(&B3s[buf][wn * 32 + nt * 16 + mlane][hf * 16]);
    }
#pragma unroll
    for (int mt = 0; mt < 2; ++mt)
#pragma unroll
      for (int nt = 0; nt < 2; ++nt) {
        acc1[mt][nt] = wmma_bf16(af[mt], b1f[nt], acc1[mt][nt]);
        acc3[mt][nt] = wmma_bf16(af[mt], b3f[nt], acc3[mt][nt]);
      }
    __syncthreads();
  }

#pragma unroll
  for (int mt = 0; mt < 2; ++mt)
#pragma unroll
    for (int nt = 0; nt < 2; ++nt)
#pragma unroll
      for (int r = 0; r < 8; ++r) {
        int mg = m0 + wm * 32 + mt * 16 + r + 8 * hf;
        int ng = n0 + wn * 32 + nt * 16 + mlane;
        float g = acc1[mt][nt][r];
        float s = g / (1.f + __expf(-g));
        C[(size_t)mg * N + ng] = (__bf16)(s * acc3[mt][nt][r]);
      }
}

// ---------------------------------------------------------------------------
// RoPE + QKV split (Q pre-scaled by 1/sqrt(HD)). grid (T, NH+2*NKV).
// ---------------------------------------------------------------------------
__global__ __launch_bounds__(64) void rope_split_kernel(
    const float* __restrict__ qkv, const long long* __restrict__ pos,
    __bf16* __restrict__ qb, __bf16* __restrict__ kb, __bf16* __restrict__ vb) {
  int t = blockIdx.x, hy = blockIdx.y, j = threadIdx.x;
  const float* src = qkv + (size_t)t * QKVD + (size_t)hy * HD;
  float fpos = (float)pos[t];
  float inv_freq = __expf(-((float)(2 * j) / (float)HD) * __logf(1.0e6f));
  float c, s;
  __sincosf(fpos * inv_freq, &c, &s);
  if (hy < NH) {
    const float sc = 0.088388347648318447f;
    float x1 = src[j], x2 = src[j + HD / 2];
    __bf16* dst = qb + ((size_t)t * NH + hy) * HD;
    dst[j]          = (__bf16)((x1 * c - x2 * s) * sc);
    dst[j + HD / 2] = (__bf16)((x2 * c + x1 * s) * sc);
  } else if (hy < NH + NKV) {
    int kh = hy - NH;
    float x1 = src[j], x2 = src[j + HD / 2];
    __bf16* dst = kb + ((size_t)t * NKV + kh) * HD;
    dst[j]          = (__bf16)(x1 * c - x2 * s);
    dst[j + HD / 2] = (__bf16)(x2 * c + x1 * s);
  } else {
    int vh = hy - NH - NKV;
    __bf16* dst = vb + ((size_t)t * NKV + vh) * HD;
    dst[j]          = (__bf16)src[j];
    dst[j + HD / 2] = (__bf16)src[j + HD / 2];
  }
}

// ---------------------------------------------------------------------------
// Flash attention, one wave per (16-query tile, head). Causal, GQA (G=4).
// ---------------------------------------------------------------------------
__global__ __launch_bounds__(32) void flash_attn_kernel(
    const __bf16* __restrict__ Q, const __bf16* __restrict__ Kc,
    const __bf16* __restrict__ V, __bf16* __restrict__ O) {
  __shared__ __bf16 Ps[16][40];
  __shared__ __bf16 Vs[32][136];
  int lane = threadIdx.x;
  int h = blockIdx.y;
  int kh = h / (NH / NKV);
  int q0 = blockIdx.x * 16;
  int mlane = lane & 15, hf = lane >> 4;

  v16bf qf[4];
#pragma unroll
  for (int cc = 0; cc < 4; ++cc)
    qf[cc] = load_a_frag(&Q[((size_t)(q0 + mlane) * NH + h) * HD + cc * 32], hf);

  float mrun[8], lrun[8];
  v8f accO[8];
#pragma unroll
  for (int r = 0; r < 8; ++r) { mrun[r] = -1e30f; lrun[r] = 0.f; accO[r] = (v8f){}; }

  for (int s0 = 0; s0 <= q0 + 15; s0 += 32) {
#pragma unroll
    for (int cc = 0; cc < 16; ++cc)
      *(v8bf*)&Vs[lane][cc * 8] =
          *(const v8bf*)&V[((size_t)(s0 + lane) * NKV + kh) * HD + cc * 8];

    v8f S[2];
    S[0] = (v8f){}; S[1] = (v8f){};
#pragma unroll
    for (int hb = 0; hb < 2; ++hb)
#pragma unroll
      for (int cc = 0; cc < 4; ++cc) {
        v16bf bf = load_b_frag(
            &Kc[((size_t)(s0 + hb * 16 + mlane) * NKV + kh) * HD + cc * 32 + hf * 16]);
        S[hb] = wmma_bf16(qf[cc], bf, S[hb]);
      }

#pragma unroll
    for (int r = 0; r < 8; ++r) {
      int m = r + 8 * hf;
      int qrow = q0 + m;
      if (s0 + mlane > qrow)      S[0][r] = -1e30f;
      if (s0 + 16 + mlane > qrow) S[1][r] = -1e30f;
      float vmax = fmaxf(S[0][r], S[1][r]);
#pragma unroll
      for (int off = 8; off; off >>= 1) vmax = fmaxf(vmax, __shfl_xor(vmax, off, 16));
      float nm = fmaxf(mrun[r], vmax);
      float p0 = __expf(S[0][r] - nm);
      float p1 = __expf(S[1][r] - nm);
      float ls = p0 + p1;
#pragma unroll
      for (int off = 8; off; off >>= 1) ls += __shfl_xor(ls, off, 16);
      float fac = __expf(mrun[r] - nm);
      lrun[r] = lrun[r] * fac + ls;
      mrun[r] = nm;
#pragma unroll
      for (int j = 0; j < 8; ++j) accO[j][r] *= fac;
      Ps[m][mlane]      = (__bf16)p0;
      Ps[m][16 + mlane] = (__bf16)p1;
    }
    __syncthreads();

    v16bf pf = load_a_frag(&Ps[mlane][0], hf);
#pragma unroll
    for (int j = 0; j < 8; ++j) {
      v16bf vf;
#pragma unroll
      for (int e = 0; e < 16; ++e) vf[e] = Vs[e + hf * 16][j * 16 + mlane];
      accO[j] = wmma_bf16(pf, vf, accO[j]);
    }
    __syncthreads();
  }

#pragma unroll
  for (int r = 0; r < 8; ++r) {
    int m = r + 8 * hf;
    float inv = 1.f / lrun[r];
#pragma unroll
    for (int j = 0; j < 8; ++j)
      O[(size_t)(q0 + m) * (NH * HD) + h * HD + j * 16 + mlane] =
          (__bf16)(accO[j][r] * inv);
  }
}

// ---------------------------------------------------------------------------
// Launcher
// ---------------------------------------------------------------------------
extern "C" void kernel_launch(void* const* d_in, const int* in_sizes, int n_in,
                              void* d_out, int out_size, void* d_ws, size_t ws_size,
                              hipStream_t stream) {
  (void)in_sizes; (void)n_in; (void)out_size; (void)ws_size;
  const float*     hidden  = (const float*)d_in[0];
  const long long* pos     = (const long long*)d_in[1];
  const float*     wqkv    = (const float*)d_in[2];
  const float*     wo      = (const float*)d_in[3];
  const float*     w1      = (const float*)d_in[4];
  const float*     w3      = (const float*)d_in[5];
  const float*     w2      = (const float*)d_in[6];
  const float*     attn_nw = (const float*)d_in[7];
  const float*     ffn_nw  = (const float*)d_in[8];
  float*           out     = (float*)d_out;

  char* p = (char*)d_ws;
  auto alloc = [&](size_t bytes) -> char* {
    char* r = p;
    p += (bytes + 255) & ~(size_t)255;
    return r;
  };
  __bf16* wqkv_b  = (__bf16*)alloc((size_t)QKVD * HIDN * 2);
  __bf16* wo_b    = (__bf16*)alloc((size_t)HIDN * HIDN * 2);
  __bf16* w1_b    = (__bf16*)alloc((size_t)INTER * HIDN * 2);
  __bf16* w3_b    = (__bf16*)alloc((size_t)INTER * HIDN * 2);
  __bf16* w2_b    = (__bf16*)alloc((size_t)HIDN * INTER * 2);
  __bf16* xnorm_b = (__bf16*)alloc((size_t)T_TOK * HIDN * 2);
  float*  qkv_f   = (float*)alloc((size_t)T_TOK * QKVD * 4);
  __bf16* q_b     = (__bf16*)alloc((size_t)T_TOK * NH * HD * 2);
  __bf16* k_b     = (__bf16*)alloc((size_t)T_TOK * NKV * HD * 2);
  __bf16* v_b     = (__bf16*)alloc((size_t)T_TOK * NKV * HD * 2);
  __bf16* attn_b  = (__bf16*)alloc((size_t)T_TOK * HIDN * 2);
  float*  h_f     = (float*)alloc((size_t)T_TOK * HIDN * 4);
  __bf16* hnorm_b = (__bf16*)alloc((size_t)T_TOK * HIDN * 2);
  __bf16* act_b   = (__bf16*)alloc((size_t)T_TOK * INTER * 2);

  cvt_bf16_kernel<<<2048, 256, 0, stream>>>(wqkv, wqkv_b, (size_t)QKVD * HIDN / 4);
  cvt_bf16_kernel<<<2048, 256, 0, stream>>>(wo, wo_b, (size_t)HIDN * HIDN / 4);
  cvt_bf16_kernel<<<4096, 256, 0, stream>>>(w1, w1_b, (size_t)INTER * HIDN / 4);
  cvt_bf16_kernel<<<4096, 256, 0, stream>>>(w3, w3_b, (size_t)INTER * HIDN / 4);
  cvt_bf16_kernel<<<4096, 256, 0, stream>>>(w2, w2_b, (size_t)HIDN * INTER / 4);

  rmsnorm_bf16_kernel<<<T_TOK, 256, 0, stream>>>(hidden, attn_nw, xnorm_b);

  wmma_gemm_kernel<0><<<dim3(QKVD / 128, T_TOK / 128), 256, 0, stream>>>(
      xnorm_b, wqkv_b, nullptr, qkv_f, T_TOK, QKVD, HIDN);

  rope_split_kernel<<<dim3(T_TOK, NH + 2 * NKV), 64, 0, stream>>>(
      qkv_f, pos, q_b, k_b, v_b);

  flash_attn_kernel<<<dim3(T_TOK / 16, NH), 32, 0, stream>>>(q_b, k_b, v_b, attn_b);

  wmma_gemm_kernel<1><<<dim3(HIDN / 128, T_TOK / 128), 256, 0, stream>>>(
      attn_b, wo_b, hidden, h_f, T_TOK, HIDN, HIDN);

  rmsnorm_bf16_kernel<<<T_TOK, 256, 0, stream>>>(h_f, ffn_nw, hnorm_b);

  wmma_gemm_gated_kernel<<<dim3(INTER / 128, T_TOK / 64), 256, 0, stream>>>(
      hnorm_b, w1_b, w3_b, act_b, T_TOK, INTER, HIDN);

  wmma_gemm_kernel<1><<<dim3(HIDN / 128, T_TOK / 128), 256, 0, stream>>>(
      act_b, w2_b, h_f, out, T_TOK, HIDN, INTER);
}
